// Generator_89008902242462
// MI455X (gfx1250) — compile-verified, matrix operand order checked by reading, not run on hardware
//
#include <hip/hip_runtime.h>
#include <hip/hip_bf16.h>
#include <math.h>

#define C_IN   256
#define OUT_C  256
#define Bb     8
#define Hh     128
#define Ww     128
#define HWc    (Hh*Ww)
#define NT     64               // pixels per tile block (4 N-tiles of 16)
#define TPR    (Ww/NT)          // 2 tiles per image row
#define KBASE  512
#define K3X3   2304
#define XH     264              // padded half-K row (bf16): 528B, 16B aligned, conflict-free
#define PC     (NT+2)           // 66 patch columns
#define CP     136              // padded half-channel dim: 272B, 16B aligned, conflict-free
#define PATCH_ELEMS (3*PC*CP)   // 26928 bf16 = 53856 B
#define XCH_ELEMS   (NT*XH)     // 16896 bf16 = 33792 B  (fits in patch buffer)

typedef __bf16 bf16_t;
typedef __attribute__((ext_vector_type(16))) __bf16 v16bf;
typedef __attribute__((ext_vector_type(8)))  __bf16 v8bf;
typedef __attribute__((ext_vector_type(8)))  float  v8f;

__device__ __forceinline__ v16bf load_frag(const bf16_t* p0, const bf16_t* p1) {
    union U { v16bf v; v8bf h[2]; } u;
    u.h[0] = *(const v8bf*)p0;
    u.h[1] = *(const v8bf*)p1;
    return u.v;
}
__device__ __forceinline__ float sigf(float v)  { return 1.f/(1.f+__expf(-v)); }
__device__ __forceinline__ float siluf(float v) { return v*sigf(v); }
__device__ __forceinline__ float geluf(float v) { return 0.5f*v*(1.f+erff(v*0.70710678118654752f)); }
__device__ __forceinline__ float blockerf(float t, const float* CB) {
    float t2 = CB[13]*t + CB[14];   // bl_bn1 (eps 1e-4) folded
    t2 *= CB[15];                   // bl_w1*bl_w2
    return CB[16]*t2 + CB[17];      // bl_bn2 folded
}

// ---------------- small prep kernels ----------------

__global__ __launch_bounds__(256) void mean_k(const float* __restrict__ x, float* __restrict__ S) {
    __shared__ float red[256];
    int bc = blockIdx.x;
    const float* p = x + (size_t)bc*HWc;
    float s = 0.f;
    for (int i = threadIdx.x; i < HWc; i += 256) s += p[i];
    red[threadIdx.x] = s; __syncthreads();
    for (int st = 128; st; st >>= 1) { if (threadIdx.x < st) red[threadIdx.x] += red[threadIdx.x+st]; __syncthreads(); }
    if (threadIdx.x == 0) S[bc] = red[0] * (1.f/(float)HWc);
}

__global__ __launch_bounds__(256) void se_k(const float* __restrict__ S,
        const float* __restrict__ w1, const float* __restrict__ b1,
        const float* __restrict__ w2, const float* __restrict__ b2,
        float* __restrict__ SCALE) {
    __shared__ float ss[256];
    __shared__ float zz[16];
    int b = blockIdx.x, t = threadIdx.x;
    ss[t] = S[b*256 + t]; __syncthreads();
    if (t < 16) {
        float a = b1[t];
        for (int c = 0; c < 256; ++c) a += w1[t*256 + c]*ss[c];
        zz[t] = a > 0.f ? a : 0.f;
    }
    __syncthreads();
    float a = b2[t];
    for (int j = 0; j < 16; ++j) a += w2[t*16 + j]*zz[j];
    SCALE[b*256 + t] = 1.f + sigf(a);   // weight_query = x*(g+1)
}

__global__ void fold_bn256_k(const float* g, const float* b, const float* m, const float* v,
                             float eps, float* A, float* B) {
    int t = threadIdx.x;
    float a = g[t]*rsqrtf(v[t]+eps);
    A[t] = a; B[t] = b[t] - m[t]*a;
}

__global__ void fold_bn1_k(const float* g, const float* b, const float* m, const float* v,
                           float eps, float* out2) {
    float a = g[0]*rsqrtf(v[0]+eps);
    out2[0] = a; out2[1] = b[0] - m[0]*a;
}

__global__ void misc_k(const float* wc1, const float* wcb,
                       const float* blw1, const float* blw2,
                       const float* cellp, const float* alpha,
                       const float* igb, const float* ogb, const float* ctb, const float* cgb,
                       const float* otw, const float* otb,
                       float* WEFF, float* CB) {
    int t = threadIdx.x;
    if (t < 256) WEFF[t] = wc1[t] + wc1[256 + t];
    if (t == 0) {
        CB[0]  = igb[0]; CB[3]  = ogb[0]; CB[6] = ctb[0];
        CB[9]  = otw[0]; CB[10] = otb[0];
        CB[15] = blw1[0]*blw2[0];
        CB[18] = wcb[0] + wcb[1];
        CB[19] = alpha[0]; CB[20] = cellp[0]; CB[23] = cgb[0];
    }
}

__global__ __launch_bounds__(256) void convw_base_k(const float* __restrict__ w, bf16_t* __restrict__ o) {
    int r = blockIdx.x, t = threadIdx.x;
    for (int k = t; k < KBASE; k += 256) o[(size_t)r*KBASE + k] = (bf16_t)w[(size_t)r*KBASE + k];
}

__global__ __launch_bounds__(256) void convw_3x3_k(const float* __restrict__ w, bf16_t* __restrict__ o) {
    int oc = blockIdx.x, c = threadIdx.x;
    for (int tap = 0; tap < 9; ++tap)
        o[(size_t)oc*K3X3 + tap*256 + c] = (bf16_t)w[((size_t)oc*256 + c)*9 + tap];
}

// ---------------- pass A: bg channel partial sums (WMMA base GEMM, N=64) ----------------

__global__ __launch_bounds__(256) void bg_sums_k(const float* __restrict__ x, const float* __restrict__ y,
        const bf16_t* __restrict__ bwB,
        const float* __restrict__ A1c, const float* __restrict__ B1c,
        const float* __restrict__ A2c, const float* __restrict__ B2c,
        const float* __restrict__ CB, float* __restrict__ PART) {
    __shared__ __align__(16) bf16_t sX[XCH_ELEMS];
    __shared__ float sSum[256];
    int tid = threadIdx.x, lane = tid & 31, wv = tid >> 5;
    int bid = blockIdx.x;
    int b = bid / (Hh*TPR); int rem = bid % (Hh*TPR);
    int h = rem / TPR; int w0 = (rem % TPR)*NT;
    float cellp = CB[20];
    int m = lane & 15, koff = (lane >> 4)*8, hl = lane >> 4;

    v8f zero8 = {0.f,0.f,0.f,0.f,0.f,0.f,0.f,0.f};
    v8f accB[2][4];
    #pragma unroll
    for (int mi = 0; mi < 2; ++mi)
        #pragma unroll
        for (int nt = 0; nt < 4; ++nt) accB[mi][nt] = zero8;

    for (int kh = 0; kh < 2; ++kh) {
        __syncthreads();
        const float* src = kh ? y : x;
        for (int idx = tid; idx < NT*256; idx += 256) {
            int n = idx & (NT-1); int k = idx >> 6;
            float v = src[(((size_t)b*256 + k)*Hh + h)*Ww + w0 + n] * cellp;
            sX[n*XH + k] = (bf16_t)v;
        }
        __syncthreads();
        #pragma unroll
        for (int mi = 0; mi < 2; ++mi) {
            int mt = wv + 8*mi;
            const bf16_t* arow = bwB + (size_t)(mt*16 + m)*KBASE + kh*256;
            #pragma unroll
            for (int k0 = 0; k0 < 256; k0 += 32) {
                v16bf af = load_frag(arow + k0 + koff, arow + k0 + koff + 16);
                #pragma unroll
                for (int nt = 0; nt < 4; ++nt) {
                    const bf16_t* bp = sX + (nt*16 + m)*XH + k0 + koff;
                    v16bf bfv = load_frag(bp, bp + 16);
                    accB[mi][nt] = __builtin_amdgcn_wmma_f32_16x16x32_bf16(false, af, false, bfv, (short)0, accB[mi][nt], false, false);
                }
            }
        }
    }
    // epilogue: bg nonlinearity, deterministic reduce over the 64 pixels
    #pragma unroll
    for (int mi = 0; mi < 2; ++mi) {
        int obase = (wv + 8*mi)*16 + (hl << 3);
        #pragma unroll
        for (int r = 0; r < 8; ++r) {
            int o = obase + r;
            float a1 = A1c[o], b1 = B1c[o], a2 = A2c[o], b2 = B2c[o];
            float s = 0.f;
            #pragma unroll
            for (int nt = 0; nt < 4; ++nt)
                s += a2*sigf(a1*accB[mi][nt][r] + b1) + b2;
            s += __shfl_xor(s, 1, 32); s += __shfl_xor(s, 2, 32);
            s += __shfl_xor(s, 4, 32); s += __shfl_xor(s, 8, 32);
            if ((lane & 15) == 0) sSum[o] = s;
        }
    }
    __syncthreads();
    PART[(size_t)bid*256 + tid] = sSum[tid];
}

__global__ __launch_bounds__(256) void cell_gate_k(const float* __restrict__ PART,
        const float* __restrict__ cgw, const float* __restrict__ CB, float* __restrict__ cgate) {
    __shared__ float red[256];
    int b = blockIdx.x, o = threadIdx.x;
    const float* p = PART + (size_t)b*(Hh*TPR)*256 + o;
    float s = 0.f;
    for (int t = 0; t < Hh*TPR; ++t) s += p[(size_t)t*256];
    red[o] = cgw[o] * (s * (1.f/(float)HWc));
    __syncthreads();
    for (int st = 128; st; st >>= 1) { if (o < st) red[o] += red[o+st]; __syncthreads(); }
    if (o == 0) {
        float v = CB[21]*(red[0] + CB[23]) + CB[22];
        cgate[b] = siluf(v);
    }
}

// ---------------- pass B: fused dual-GEMM (N=64, register-blocked) + gate chain ----------------

__global__ __launch_bounds__(256) void fused_main_k(const float* __restrict__ x, const float* __restrict__ y,
        const bf16_t* __restrict__ bwB, const bf16_t* __restrict__ w3B,
        const float* __restrict__ scaleBC, const float* __restrict__ cgate,
        const float* __restrict__ A1c, const float* __restrict__ B1c,
        const float* __restrict__ A2c, const float* __restrict__ B2c,
        const float* __restrict__ WEFF, const float* __restrict__ IGW,
        const float* __restrict__ OGW, const float* __restrict__ CTW,
        const float* __restrict__ CB,
        float* __restrict__ out_fuse, float* __restrict__ out_ws) {
    // union buffer: base-GEMM K-half tile OR conv3x3 patch half (phases are barrier-separated)
    __shared__ __align__(16) bf16_t sBuf[PATCH_ELEMS];
    __shared__ float sSc[256];
    __shared__ float sPart[4][8][NT];

    int tid = threadIdx.x, lane = tid & 31, wv = tid >> 5;
    int bid = blockIdx.x;
    int b = bid / (Hh*TPR); int rem = bid % (Hh*TPR);
    int h = rem / TPR; int w0 = (rem % TPR)*NT;
    float cellp = CB[20];
    sSc[tid] = scaleBC[b*256 + tid];

    int m = lane & 15, koff = (lane >> 4)*8, hl = lane >> 4;
    v8f zero8 = {0.f,0.f,0.f,0.f,0.f,0.f,0.f,0.f};

    // ================= base conv1x1 GEMM: M=256, N=64, K=512 (two staged halves) =================
    v8f accB[2][4];
    #pragma unroll
    for (int mi = 0; mi < 2; ++mi)
        #pragma unroll
        for (int nt = 0; nt < 4; ++nt) accB[mi][nt] = zero8;

    bf16_t* sX = (bf16_t*)sBuf;
    for (int kh = 0; kh < 2; ++kh) {
        __syncthreads();
        const float* src = kh ? y : x;
        for (int idx = tid; idx < NT*256; idx += 256) {
            int n = idx & (NT-1); int k = idx >> 6;
            float v = src[(((size_t)b*256 + k)*Hh + h)*Ww + w0 + n] * cellp;
            sX[n*XH + k] = (bf16_t)v;
        }
        __syncthreads();
        // prefetch the second M-tile's weight row while the first computes
        __builtin_prefetch((const void*)(bwB + (size_t)((wv + 8)*16 + m)*KBASE + kh*256 + koff), 0, 1);
        #pragma unroll
        for (int mi = 0; mi < 2; ++mi) {
            int mt = wv + 8*mi;
            const bf16_t* arow = bwB + (size_t)(mt*16 + m)*KBASE + kh*256;
            #pragma unroll
            for (int k0 = 0; k0 < 256; k0 += 32) {
                v16bf af = load_frag(arow + k0 + koff, arow + k0 + koff + 16);
                #pragma unroll
                for (int nt = 0; nt < 4; ++nt) {
                    const bf16_t* bp = sX + (nt*16 + m)*XH + k0 + koff;
                    v16bf bfv = load_frag(bp, bp + 16);
                    accB[mi][nt] = __builtin_amdgcn_wmma_f32_16x16x32_bf16(false, af, false, bfv, (short)0, accB[mi][nt], false, false);
                }
            }
        }
    }
    // base epilogue: BN/sigmoid/BN then channel-dots, halves combined via lane-16 shuffle
    {
        float pIG[4] = {0,0,0,0}, pOG[4] = {0,0,0,0}, pCT[4] = {0,0,0,0};
        #pragma unroll
        for (int mi = 0; mi < 2; ++mi) {
            int obase = (wv + 8*mi)*16 + (hl << 3);
            #pragma unroll
            for (int r = 0; r < 8; ++r) {
                int o = obase + r;
                float a1 = A1c[o], b1 = B1c[o], a2 = A2c[o], b2 = B2c[o];
                float wi = IGW[o], wo = OGW[o], wc = CTW[o];
                #pragma unroll
                for (int nt = 0; nt < 4; ++nt) {
                    float bg = a2*sigf(a1*accB[mi][nt][r] + b1) + b2;
                    pIG[nt] += wi*bg; pOG[nt] += wo*bg; pCT[nt] += wc*bg;
                }
            }
        }
        #pragma unroll
        for (int nt = 0; nt < 4; ++nt) {
            float vI = pIG[nt] + __shfl_xor(pIG[nt], 16, 32);
            float vO = pOG[nt] + __shfl_xor(pOG[nt], 16, 32);
            float vC = pCT[nt] + __shfl_xor(pCT[nt], 16, 32);
            if (hl == 0) {
                sPart[0][wv][nt*16 + m] = vI;
                sPart[1][wv][nt*16 + m] = vO;
                sPart[2][wv][nt*16 + m] = vC;
            }
        }
    }

    // ================= conv3x3 implicit GEMM: M=256, N=64, K=9x256 (two channel halves) =================
    v8f accC[2][4];
    #pragma unroll
    for (int mi = 0; mi < 2; ++mi)
        #pragma unroll
        for (int nt = 0; nt < 4; ++nt) accC[mi][nt] = zero8;

    bf16_t* sP = (bf16_t*)sBuf;
    for (int pass = 0; pass < 2; ++pass) {
        __syncthreads();
        // stage weight_query patch (x*scale) for channels [pass*128, pass*128+128)
        for (int chb = 0; chb < 16; ++chb) {
            int chl = chb*8 + wv;
            float sc = sSc[pass*128 + chl];
            #pragma unroll
            for (int r = 0; r < 3; ++r) {
                int hr = h - 1 + r;
                bool rok = (hr >= 0) && (hr < Hh);
                const float* rowp = x + (((size_t)b*256 + pass*128 + chl)*Hh + (rok ? hr : 0))*Ww;
                #pragma unroll
                for (int cp2 = 0; cp2 < 3; ++cp2) {
                    int col = lane + 32*cp2;
                    if (col < PC) {
                        int wc2 = w0 - 1 + col;
                        float v = 0.f;
                        if (rok && wc2 >= 0 && wc2 < Ww) v = rowp[wc2]*sc;
                        sP[(r*PC + col)*CP + chl] = (bf16_t)v;
                    }
                }
            }
        }
        __syncthreads();
        __builtin_prefetch((const void*)(w3B + (size_t)((wv + 8)*16 + m)*K3X3 + pass*128 + koff), 0, 1);
        #pragma unroll
        for (int mi = 0; mi < 2; ++mi) {
            int mt = wv + 8*mi;
            const bf16_t* arow3 = w3B + (size_t)(mt*16 + m)*K3X3 + pass*128;
            for (int tap = 0; tap < 9; ++tap) {
                int dy = tap/3, dx = tap%3;
                const bf16_t* ap = arow3 + tap*256;
                #pragma unroll
                for (int kc = 0; kc < 128; kc += 32) {
                    v16bf af = load_frag(ap + kc + koff, ap + kc + koff + 16);
                    #pragma unroll
                    for (int nt = 0; nt < 4; ++nt) {
                        const bf16_t* bp = sP + ((dy*PC) + (nt*16 + m + dx))*CP + kc + koff;
                        v16bf bfv = load_frag(bp, bp + 16);
                        accC[mi][nt] = __builtin_amdgcn_wmma_f32_16x16x32_bf16(false, af, false, bfv, (short)0, accC[mi][nt], false, false);
                    }
                }
            }
        }
    }
    // conv3x3 epilogue: ReLU then w_eff dot
    {
        float pWS[4] = {0,0,0,0};
        #pragma unroll
        for (int mi = 0; mi < 2; ++mi) {
            int obase = (wv + 8*mi)*16 + (hl << 3);
            #pragma unroll
            for (int r = 0; r < 8; ++r) {
                float we = WEFF[obase + r];
                #pragma unroll
                for (int nt = 0; nt < 4; ++nt)
                    pWS[nt] += we*fmaxf(accC[mi][nt][r], 0.f);
            }
        }
        #pragma unroll
        for (int nt = 0; nt < 4; ++nt) {
            float vW = pWS[nt] + __shfl_xor(pWS[nt], 16, 32);
            if (hl == 0) sPart[3][wv][nt*16 + m] = vW;
        }
    }
    __syncthreads();

    // ================= per-pixel gate chain =================
    if (tid < NT) {
        int n = tid;
        float aIG = 0.f, aOG = 0.f, aCT = 0.f, aWS = 0.f;
        #pragma unroll
        for (int s = 0; s < 8; ++s) {
            aIG += sPart[0][s][n]; aOG += sPart[1][s][n];
            aCT += sPart[2][s][n]; aWS += sPart[3][s][n];
        }
        float wsv  = aWS + CB[18];                                    // weight_soft
        float igv  = siluf(CB[1]*(aIG + CB[0]) + CB[2]);              // input_gate
        float ogv  = siluf(CB[4]*(aOG + CB[3]) + CB[5]) + CB[19]*igv; // output_gate
        float ctv  = geluf(CB[7]*(aCT + CB[6]) + CB[8]);              // cell_task
        float cellv = cgate[b]*cellp + igv*ctv;
        cellv = blockerf(cellv, CB) + cellv;
        float otv  = geluf(CB[11]*(CB[9]*cellv + CB[10]) + CB[12]);   // out_task
        float fuse = blockerf(otv*ogv*wsv, CB) * 0.4f;                // BETA
        size_t oidx = (((size_t)b*Hh) + h)*Ww + w0 + n;
        out_fuse[oidx] = fuse;
        out_ws[oidx]   = wsv;
    }
}

// ---------------- host launch ----------------

extern "C" void kernel_launch(void* const* d_in, const int* in_sizes, int n_in,
                              void* d_out, int out_size, void* d_ws, size_t ws_size,
                              hipStream_t stream) {
    (void)in_sizes; (void)n_in; (void)out_size; (void)ws_size;
    const float* x      = (const float*)d_in[0];
    const float* y      = (const float*)d_in[1];
    const float* se_w1  = (const float*)d_in[2];
    const float* se_b1  = (const float*)d_in[3];
    const float* se_w2  = (const float*)d_in[4];
    const float* se_b2  = (const float*)d_in[5];
    const float* base_w = (const float*)d_in[6];
    const float *bn1g=(const float*)d_in[7],  *bn1b=(const float*)d_in[8],  *bn1m=(const float*)d_in[9],  *bn1v=(const float*)d_in[10];
    const float *bn2g=(const float*)d_in[11], *bn2b=(const float*)d_in[12], *bn2m=(const float*)d_in[13], *bn2v=(const float*)d_in[14];
    const float* ig_w=(const float*)d_in[15]; const float* ig_b=(const float*)d_in[16];
    const float *iga=(const float*)d_in[17], *igb2=(const float*)d_in[18], *igm=(const float*)d_in[19], *igv=(const float*)d_in[20];
    const float* og_w=(const float*)d_in[21]; const float* og_b=(const float*)d_in[22];
    const float *oga=(const float*)d_in[23], *ogb2=(const float*)d_in[24], *ogm=(const float*)d_in[25], *ogv=(const float*)d_in[26];
    const float* cg_w=(const float*)d_in[27]; const float* cg_b=(const float*)d_in[28];
    const float *cga=(const float*)d_in[29], *cgb2=(const float*)d_in[30], *cgm=(const float*)d_in[31], *cgv=(const float*)d_in[32];
    const float* ct_w=(const float*)d_in[33]; const float* ct_b=(const float*)d_in[34];
    const float *cta=(const float*)d_in[35], *ctb2=(const float*)d_in[36], *ctm=(const float*)d_in[37], *ctv=(const float*)d_in[38];
    const float* ot_w=(const float*)d_in[39]; const float* ot_b=(const float*)d_in[40];
    const float *ota=(const float*)d_in[41], *otb2=(const float*)d_in[42], *otm=(const float*)d_in[43], *otv=(const float*)d_in[44];
    const float* wc_w3=(const float*)d_in[45];
    const float* wc_w1=(const float*)d_in[46];
    const float* wc_b1=(const float*)d_in[47];
    const float *bl1g=(const float*)d_in[48], *bl1b=(const float*)d_in[49], *bl1m=(const float*)d_in[50], *bl1v=(const float*)d_in[51];
    const float* bl_w1=(const float*)d_in[52];
    const float* bl_w2=(const float*)d_in[53];
    const float *bl2g=(const float*)d_in[54], *bl2b=(const float*)d_in[55], *bl2m=(const float*)d_in[56], *bl2v=(const float*)d_in[57];
    const float* cellp=(const float*)d_in[58];
    const float* alpha=(const float*)d_in[59];

    // workspace carve (16B-aligned for bf16 b128 fragment loads)
    float* S     = (float*)d_ws;          // 2048
    float* SCALE = S + 2048;              // 2048
    float* CGATE = SCALE + 2048;          // 32 (8 used)
    float* A1    = CGATE + 32;            // 256 each
    float* B1    = A1 + 256;
    float* A2    = B1 + 256;
    float* B2    = A2 + 256;
    float* WEFF  = B2 + 256;
    float* CB    = WEFF + 256;            // 32 scalars
    bf16_t* BWB  = (bf16_t*)(CB + 32);                 // 256*512  bf16
    bf16_t* W3B  = BWB + (size_t)OUT_C*KBASE;          // 256*2304 bf16
    float* PART  = (float*)(W3B + (size_t)OUT_C*K3X3); // 2048*256 floats

    const int nTiles = Bb*Hh*TPR;   // 2048

    mean_k<<<Bb*C_IN, 256, 0, stream>>>(x, S);
    se_k<<<Bb, 256, 0, stream>>>(S, se_w1, se_b1, se_w2, se_b2, SCALE);
    fold_bn256_k<<<1, 256, 0, stream>>>(bn1g, bn1b, bn1m, bn1v, 1e-5f, A1, B1);
    fold_bn256_k<<<1, 256, 0, stream>>>(bn2g, bn2b, bn2m, bn2v, 1e-5f, A2, B2);
    fold_bn1_k<<<1, 1, 0, stream>>>(iga, igb2, igm, igv, 1e-5f, CB + 1);
    fold_bn1_k<<<1, 1, 0, stream>>>(oga, ogb2, ogm, ogv, 1e-5f, CB + 4);
    fold_bn1_k<<<1, 1, 0, stream>>>(cta, ctb2, ctm, ctv, 1e-5f, CB + 7);
    fold_bn1_k<<<1, 1, 0, stream>>>(ota, otb2, otm, otv, 1e-5f, CB + 11);
    fold_bn1_k<<<1, 1, 0, stream>>>(bl1g, bl1b, bl1m, bl1v, 1e-4f, CB + 13);
    fold_bn1_k<<<1, 1, 0, stream>>>(bl2g, bl2b, bl2m, bl2v, 1e-4f, CB + 16);
    fold_bn1_k<<<1, 1, 0, stream>>>(cga, cgb2, cgm, cgv, 1e-5f, CB + 21);
    misc_k<<<1, 256, 0, stream>>>(wc_w1, wc_b1, bl_w1, bl_w2, cellp, alpha,
                                  ig_b, og_b, ct_b, cg_b, ot_w, ot_b, WEFF, CB);
    convw_base_k<<<OUT_C, 256, 0, stream>>>(base_w, BWB);
    convw_3x3_k<<<OUT_C, 256, 0, stream>>>(wc_w3, W3B);

    bg_sums_k<<<nTiles, 256, 0, stream>>>(x, y, BWB, A1, B1, A2, B2, CB, PART);
    cell_gate_k<<<Bb, 256, 0, stream>>>(PART, cg_w, CB, CGATE);

    float* out_fuse = (float*)d_out;
    float* out_ws2  = out_fuse + (size_t)Bb*HWc;
    fused_main_k<<<nTiles, 256, 0, stream>>>(x, y, BWB, W3B, SCALE, CGATE,
                                             A1, B1, A2, B2, WEFF, ig_w, og_w, ct_w,
                                             CB, out_fuse, out_ws2);
}
